// TransformerDecoderLayer_16638703305118
// MI455X (gfx1250) — matrix operand order repaired and checked by reading, not compiled, not run on hardware
//
#include <hip/hip_runtime.h>
#include <stdint.h>
#include <math.h>

// ---------------------------------------------------------------------------
// Types for CDNA5 WMMA (wave32): v16bf A/B fragments, v8f accumulator.
// ---------------------------------------------------------------------------
typedef __attribute__((ext_vector_type(16))) __bf16       v16bf;
typedef __attribute__((ext_vector_type(8)))  float        v8f;
typedef __attribute__((ext_vector_type(4)))  unsigned int v4u;

union Frag { v4u h[2]; v16bf v; };

__device__ __forceinline__ unsigned short f2bf(float f) {
    union { float f; unsigned int u; } c; c.f = f;
    unsigned int u = c.u;
    return (unsigned short)((u + 0x7FFFu + ((u >> 16) & 1u)) >> 16); // RNE
}

// CDNA5 async global->LDS copy, 16B per lane (tracked by ASYNCcnt).
// GLOBAL_LOAD_ASYNC_TO_LDS_B128, GV mode.  LDS offset = low 32 bits of the
// generic __shared__ address (LDS aperture => addr[31:0] is the offset).
__device__ __forceinline__ void async_ld128(unsigned lds_off, const void* gaddr) {
    asm volatile("global_load_async_to_lds_b128 %0, %1, off"
                 :: "v"(lds_off), "v"(gaddr) : "memory");
}
// Per-wave async loads complete in order: waiting ASYNCcnt<=6 after issuing
// the next tile's 6 ops guarantees the current tile's 6 ops have landed.
__device__ __forceinline__ void wait_async6() {
    asm volatile("s_wait_asynccnt 0x6" ::: "memory");
}
__device__ __forceinline__ void wait_async0() {
    asm volatile("s_wait_asynccnt 0x0" ::: "memory");
}

// ---------------------------------------------------------------------------
// f32 -> bf16 elementwise convert
// ---------------------------------------------------------------------------
__global__ __launch_bounds__(256)
void cvt_f32_bf16(const float* __restrict__ in, unsigned short* __restrict__ out, size_t n) {
    size_t i = (size_t)blockIdx.x * 256u + threadIdx.x;
    size_t stride = (size_t)gridDim.x * 256u;
    for (; i < n; i += stride) out[i] = f2bf(in[i]);
}

// ---------------------------------------------------------------------------
// Tiled transpose + convert: out[c][r] = bf16(in[r][c]).  32x32 tiles via LDS.
// Used for score_c and all weight matrices so every GEMM B operand is stored
// [N][K] (fast 16B-coalesced async tile loads, no scalar LDS stores).
// ---------------------------------------------------------------------------
__global__ __launch_bounds__(256)
void transpose_cvt(const float* __restrict__ in, unsigned short* __restrict__ out,
                   int R, int C, size_t sIn, size_t sOut) {
    __shared__ float tile[32][33];
    const float* src = in + (size_t)blockIdx.z * sIn;
    unsigned short* dst = out + (size_t)blockIdx.z * sOut;
    int c  = threadIdx.x & 31;
    int r0 = threadIdx.x >> 5;              // 0..7
    int gr = blockIdx.y * 32, gc = blockIdx.x * 32;
#pragma unroll
    for (int i = 0; i < 4; ++i) {
        int r = r0 + i * 8;
        tile[r][c] = src[(size_t)(gr + r) * C + gc + c];
    }
    __syncthreads();
#pragma unroll
    for (int i = 0; i < 4; ++i) {
        int r = r0 + i * 8;
        dst[(size_t)(gc + r) * R + gr + c] = f2bf(tile[c][r]);
    }
}

// ---------------------------------------------------------------------------
// bf16 WMMA GEMM:  C[M][N] = alpha * A[M][K] @ B^T (+bias[n]) (opt ReLU)
//   A stored [M][K] row-major; B stored [N][K] row-major (pre-transposed).
// Block tile 64(M) x 128(N), BK=64 (two WMMA K-steps per staged tile).
// 8 waves: 2(M) x 4(N); each wave owns a 32x32 C tile = 2x2
// v_wmma_f32_16x16x32_bf16 per 32-K step.
// Double-buffered async pipeline: tile t+1 streams into LDS buffer (t+1)&1
// via GLOBAL_LOAD_ASYNC_TO_LDS_B128 while tile t is consumed from buffer t&1.
// OUT_TRANS=1 stores bf16 C transposed: C^T[n][m] with leading dim ldc(=M).
// ---------------------------------------------------------------------------
template<bool OUT_BF16, bool RELU, bool OUT_TRANS>
__global__ __launch_bounds__(256, 2)
void gemm_bf16_wmma(const unsigned short* __restrict__ A, size_t sA,
                    const unsigned short* __restrict__ B, size_t sB,
                    const float* __restrict__ bias,
                    float* __restrict__ Cf, unsigned short* __restrict__ Cb, size_t sC,
                    int M, int N, int K, int ldc, float alpha) {
    constexpr int LDA = 72;                 // padded halfword stride (144B)
    __shared__ unsigned short As[2][64 * LDA];
    __shared__ unsigned short Bs[2][128 * LDA];

    const int tid  = threadIdx.x;
    const int lane = tid & 31;
    const int wave = tid >> 5;
    const int wm   = wave >> 2;   // 0..1
    const int wn   = wave & 3;    // 0..3
    const int ml   = lane & 15;   // A row / B col within 16
    const int kh   = lane >> 4;   // K-half selector

    const size_t bz = blockIdx.z;
    A += bz * sA + (size_t)blockIdx.y * 64 * (size_t)K;
    B += bz * sB + (size_t)blockIdx.x * 128 * (size_t)K;
    if (bias) bias += (size_t)blockIdx.x * 128;

    v8f acc[2][2] = {};

    // Loader geometry: 16B chunks of 8 bf16; rows of 64 K = 8 chunks.
    const int lrow = tid >> 3;        // 0..31
    const int lkc  = (tid & 7) * 8;   // 0..56

    unsigned aL[2][2], bL[2][4];
#pragma unroll
    for (int b = 0; b < 2; ++b) {
        aL[b][0] = (unsigned)(uintptr_t)&As[b][lrow * LDA + lkc];
        aL[b][1] = (unsigned)(uintptr_t)&As[b][(32 + lrow) * LDA + lkc];
#pragma unroll
        for (int i = 0; i < 4; ++i)
            bL[b][i] = (unsigned)(uintptr_t)&Bs[b][(lrow + 32 * i) * LDA + lkc];
    }

    auto issue = [&](int buf, int k0) {
        async_ld128(aL[buf][0], &A[(size_t)lrow * K + k0 + lkc]);
        async_ld128(aL[buf][1], &A[(size_t)(32 + lrow) * K + k0 + lkc]);
#pragma unroll
        for (int i = 0; i < 4; ++i)
            async_ld128(bL[buf][i], &B[(size_t)(lrow + 32 * i) * K + k0 + lkc]);
    };

    const int nt = K >> 6;            // K / 64, exact for all shapes here
    issue(0, 0);
    for (int t = 0; t < nt; ++t) {
        const int cur = t & 1;
        if (t + 1 < nt) { issue(cur ^ 1, (t + 1) * 64); wait_async6(); }
        else            { wait_async0(); }
        __syncthreads();

#pragma unroll
        for (int ks = 0; ks < 2; ++ks) {
            Frag af[2], bf[2];
#pragma unroll
            for (int i = 0; i < 2; ++i) {
                int r = wm * 32 + i * 16 + ml;
                af[i].h[0] = *(const v4u*)&As[cur][r * LDA + ks * 32 + kh * 8];
                af[i].h[1] = *(const v4u*)&As[cur][r * LDA + ks * 32 + 16 + kh * 8];
            }
#pragma unroll
            for (int j = 0; j < 2; ++j) {
                int n = wn * 32 + j * 16 + ml;
                bf[j].h[0] = *(const v4u*)&Bs[cur][n * LDA + ks * 32 + kh * 8];
                bf[j].h[1] = *(const v4u*)&Bs[cur][n * LDA + ks * 32 + 16 + kh * 8];
            }
#pragma unroll
            for (int i = 0; i < 2; ++i)
#pragma unroll
                for (int j = 0; j < 2; ++j)
                    acc[i][j] = __builtin_amdgcn_wmma_f32_16x16x32_bf16(
                        false, af[i].v, false, bf[j].v, (short)0, acc[i][j], false, false);
        }
        __syncthreads();
    }

    const size_t cOff  = bz * sC + (size_t)blockIdx.y * 64 * (size_t)ldc
                                 + (size_t)blockIdx.x * 128;
    const size_t cOffT = bz * sC + (size_t)blockIdx.x * 128 * (size_t)ldc
                                 + (size_t)blockIdx.y * 64;
#pragma unroll
    for (int i = 0; i < 2; ++i) {
#pragma unroll
        for (int j = 0; j < 2; ++j) {
            int nloc = wn * 32 + j * 16 + ml;
            float bv = bias ? bias[nloc] : 0.0f;
#pragma unroll
            for (int r = 0; r < 8; ++r) {
                int mloc = wm * 32 + i * 16 + kh * 8 + r;
                float v = acc[i][j][r] * alpha + bv;
                if (RELU) v = fmaxf(v, 0.0f);
                if (OUT_TRANS) {
                    size_t idx = cOffT + (size_t)nloc * ldc + mloc;
                    Cb[idx] = f2bf(v);
                } else {
                    size_t idx = cOff + (size_t)mloc * ldc + nloc;
                    if (OUT_BF16) Cb[idx] = f2bf(v);
                    else          Cf[idx] = v;
                }
            }
        }
    }
}

// ---------------------------------------------------------------------------
// Block-wide reductions (wave32 shuffle + LDS across waves)
// ---------------------------------------------------------------------------
__device__ __forceinline__ float block_reduce(float v, bool isMax, float* red, int nw) {
#pragma unroll
    for (int o = 16; o > 0; o >>= 1) {
        float t = __shfl_xor(v, o, 32);
        v = isMax ? fmaxf(v, t) : (v + t);
    }
    __syncthreads();
    if ((threadIdx.x & 31) == 0) red[threadIdx.x >> 5] = v;
    __syncthreads();
    float r = red[0];
    for (int i = 1; i < nw; ++i) r = isMax ? fmaxf(r, red[i]) : (r + red[i]);
    return r;
}

// ---------------------------------------------------------------------------
// Fused dual softmax + blend over rows of length 2048:
//   score = a1*softmax(mixed) + (1-a1)*softmax(att)
// mixed lives in scoreF (d_out score region) and is overwritten in place.
// Also emits bf16 score for the z GEMM.
// ---------------------------------------------------------------------------
__global__ __launch_bounds__(256)
void softmax_blend(const float* __restrict__ att, float* __restrict__ scoreF,
                   unsigned short* __restrict__ scoreB, const float* __restrict__ a1p) {
    __shared__ float red[8];
    const size_t row = blockIdx.x;
    const float* mr = scoreF + row * 2048;
    const float* ar = att + row * 2048;
    float mv[8], av[8];
#pragma unroll
    for (int i = 0; i < 8; ++i) {
        int e = threadIdx.x + i * 256;
        mv[i] = mr[e]; av[i] = ar[e];
    }
    float mm = -3.4e38f, am = -3.4e38f;
#pragma unroll
    for (int i = 0; i < 8; ++i) { mm = fmaxf(mm, mv[i]); am = fmaxf(am, av[i]); }
    mm = block_reduce(mm, true, red, 8);
    am = block_reduce(am, true, red, 8);
    float ms = 0.f, as = 0.f;
#pragma unroll
    for (int i = 0; i < 8; ++i) {
        mv[i] = __expf(mv[i] - mm); ms += mv[i];
        av[i] = __expf(av[i] - am); as += av[i];
    }
    ms = block_reduce(ms, false, red, 8);
    as = block_reduce(as, false, red, 8);
    const float a1 = a1p[0];
    const float fm = a1 / ms, fa = (1.0f - a1) / as;
#pragma unroll
    for (int i = 0; i < 8; ++i) {
        int e = threadIdx.x + i * 256;
        float s = mv[i] * fm + av[i] * fa;
        scoreF[row * 2048 + e] = s;
        scoreB[row * 2048 + e] = f2bf(s);
    }
}

// ---------------------------------------------------------------------------
// Fused residual-add + LayerNorm over D=384 (block per row, 128 threads x 3)
// ---------------------------------------------------------------------------
__global__ __launch_bounds__(128)
void add_ln(const float* __restrict__ X, const float* __restrict__ Y,
            const float* __restrict__ g, const float* __restrict__ b,
            float* __restrict__ outF, unsigned short* __restrict__ outB) {
    __shared__ float red[4];
    const size_t row = blockIdx.x;
    float s[3];
#pragma unroll
    for (int i = 0; i < 3; ++i) {
        int e = threadIdx.x + i * 128;
        s[i] = X[row * 384 + e] + Y[row * 384 + e];
    }
    float sum = s[0] + s[1] + s[2];
    sum = block_reduce(sum, false, red, 4);
    float mean = sum * (1.0f / 384.0f);
    float var = 0.f;
#pragma unroll
    for (int i = 0; i < 3; ++i) { float d = s[i] - mean; var += d * d; }
    var = block_reduce(var, false, red, 4);
    float inv = rsqrtf(var * (1.0f / 384.0f) + 1e-5f);
#pragma unroll
    for (int i = 0; i < 3; ++i) {
        int e = threadIdx.x + i * 128;
        float y = (s[i] - mean) * inv * g[e] + b[e];
        if (outF) outF[row * 384 + e] = y;
        if (outB) outB[row * 384 + e] = f2bf(y);
    }
}

// ---------------------------------------------------------------------------
// Host-side launcher
// ---------------------------------------------------------------------------
extern "C" void kernel_launch(void* const* d_in, const int* in_sizes, int n_in,
                              void* d_out, int out_size, void* d_ws, size_t ws_size,
                              hipStream_t stream) {
    (void)in_sizes; (void)n_in; (void)out_size; (void)ws_size;

    constexpr int Bn = 8, T = 2048, D = 384, DM = 256, DF = 1536;
    const size_t BT   = (size_t)Bn * T;            // 16384 rows
    const size_t TD   = (size_t)T * D;             // 2048*384
    const size_t TM   = (size_t)T * DM;            // 2048*256
    const size_t TT   = (size_t)T * T;             // 2048*2048
    const size_t nBTD = BT * D;                    // 6291456
    const size_t nBTM = BT * DM;                   // 4194304
    const size_t nBTT = (size_t)Bn * TT;           // 33554432
    const size_t nBTF = BT * DF;                   // 25165824

    const float* tgt     = (const float*)d_in[0];
    const float* memory  = (const float*)d_in[1];
    const float* a1      = (const float*)d_in[2];
    const float* score_c = (const float*)d_in[3];
    const float* out_c   = (const float*)d_in[4];
    const float* Wq = (const float*)d_in[5];  const float* bq = (const float*)d_in[6];
    const float* Wk = (const float*)d_in[7];  const float* bk = (const float*)d_in[8];
    const float* Wv = (const float*)d_in[9];  const float* bv = (const float*)d_in[10];
    const float* Wu = (const float*)d_in[11]; const float* bu = (const float*)d_in[12];
    const float* Wn = (const float*)d_in[13]; const float* bn = (const float*)d_in[14];
    const float* W1 = (const float*)d_in[15]; const float* b1 = (const float*)d_in[16];
    const float* W2 = (const float*)d_in[17]; const float* b2 = (const float*)d_in[18];
    const float* g1 = (const float*)d_in[19]; const float* be1 = (const float*)d_in[20];
    const float* g2 = (const float*)d_in[21]; const float* be2 = (const float*)d_in[22];

    float* outF   = (float*)d_out;                 // [8,2048,384]
    float* scoreF = outF + nBTD;                   // [8,1,2048,2048]

    char* ws = (char*)d_ws; size_t off = 0;
    auto take = [&](size_t elems, size_t esz) -> void* {
        void* p = ws + off;
        off += ((elems * esz + 255) & ~(size_t)255);
        return p;
    };
    typedef unsigned short us;
    us* tgt_b  = (us*)take(nBTD, 2);
    us* mem_b  = (us*)take(nBTM, 2);
    us* outc_b = (us*)take(nBTD, 2);
    us* scT_b  = (us*)take(nBTT, 2);   // score_c^T per batch: [T1][Tc]
    us* wqT = (us*)take((size_t)D * D, 2);     // [N][K] weights
    us* wkT = (us*)take((size_t)DM * D, 2);
    us* wvT = (us*)take((size_t)DM * D, 2);
    us* wuT = (us*)take((size_t)D * D, 2);
    us* wnT = (us*)take((size_t)D * D, 2);
    us* w1T = (us*)take((size_t)D * DF, 2);
    us* w2T = (us*)take((size_t)DF * D, 2);
    us* q_b  = (us*)take(nBTD, 2);
    us* k_b  = (us*)take(nBTD, 2);
    us* vT_b = (us*)take(nBTD, 2);     // v^T per batch: [D][T1]
    us* un_b = (us*)take(nBTD, 2);
    us* kn_b = (us*)take(nBTD, 2);
    us* s1_b = (us*)take(nBTT, 2);
    us* sc_b = (us*)take(nBTT, 2);
    us* x_b  = (us*)take(nBTD, 2);
    us* h_b  = (us*)take(nBTF, 2);
    float* att_f = (float*)take(nBTT, 4);
    float* z_f   = (float*)take(nBTD, 4);
    float* x_f   = (float*)take(nBTD, 4);
    float* ff_f  = (float*)take(nBTD, 4);

    auto cvt = [&](const float* i, us* o, size_t n) {
        int blocks = (int)((n + 2047) / 2048); if (blocks > 8192) blocks = 8192;
        cvt_f32_bf16<<<dim3(blocks), dim3(256), 0, stream>>>(i, o, n);
    };
    cvt(tgt, tgt_b, nBTD);
    cvt(memory, mem_b, nBTM);
    cvt(out_c, outc_b, nBTD);
    // Weights [K][N] -> [N][K] bf16 (transpose_cvt: out[c][r] = in[r][c], in is [R][C])
    transpose_cvt<<<dim3(12, 12, 1), dim3(256), 0, stream>>>(Wq, wqT, D, D, 0, 0);
    transpose_cvt<<<dim3(12,  8, 1), dim3(256), 0, stream>>>(Wk, wkT, DM, D, 0, 0);
    transpose_cvt<<<dim3(12,  8, 1), dim3(256), 0, stream>>>(Wv, wvT, DM, D, 0, 0);
    transpose_cvt<<<dim3(12, 12, 1), dim3(256), 0, stream>>>(Wu, wuT, D, D, 0, 0);
    transpose_cvt<<<dim3(12, 12, 1), dim3(256), 0, stream>>>(Wn, wnT, D, D, 0, 0);
    transpose_cvt<<<dim3(48, 12, 1), dim3(256), 0, stream>>>(W1, w1T, D, DF, 0, 0);
    transpose_cvt<<<dim3(12, 48, 1), dim3(256), 0, stream>>>(W2, w2T, DF, D, 0, 0);
    transpose_cvt<<<dim3(64, 64, 8), dim3(256), 0, stream>>>(score_c, scT_b, T, T, TT, TT);

    const float scale = 1.0f / sqrtf((float)D);

    // Projections (A[M][K] @ W^T[N][K], bf16 out, +bias)
    gemm_bf16_wmma<true, false, false><<<dim3(3, 256, 1), 256, 0, stream>>>(
        tgt_b, 0, wqT, 0, bq, nullptr, q_b, 0, (int)BT, D, D, D, 1.0f);
    gemm_bf16_wmma<true, false, false><<<dim3(3, 256, 1), 256, 0, stream>>>(
        mem_b, 0, wkT, 0, bk, nullptr, k_b, 0, (int)BT, D, DM, D, 1.0f);
    // v projection stores v^T per batch so the z GEMM gets a [N][K] operand
    gemm_bf16_wmma<true, false, true><<<dim3(3, 32, 8), 256, 0, stream>>>(
        mem_b, TM, wvT, 0, bv, nullptr, vT_b, TD, T, D, DM, T, 1.0f);
    gemm_bf16_wmma<true, false, false><<<dim3(3, 256, 1), 256, 0, stream>>>(
        tgt_b, 0, wuT, 0, bu, nullptr, un_b, 0, (int)BT, D, D, D, 1.0f);
    gemm_bf16_wmma<true, false, false><<<dim3(3, 256, 1), 256, 0, stream>>>(
        outc_b, 0, wnT, 0, bn, nullptr, kn_b, 0, (int)BT, D, D, D, 1.0f);

    // att = scale * q @ k^T  (f32)    [8,2048,2048]
    gemm_bf16_wmma<false, false, false><<<dim3(16, 32, 8), 256, 0, stream>>>(
        q_b, TD, k_b, TD, nullptr, att_f, nullptr, TT, T, T, D, T, scale);
    // score1 = scale * unknow @ know^T  (bf16)
    gemm_bf16_wmma<true, false, false><<<dim3(16, 32, 8), 256, 0, stream>>>(
        un_b, TD, kn_b, TD, nullptr, nullptr, s1_b, TT, T, T, D, T, scale);
    // mixed = score1 @ score_c  (B pre-transposed; K=2048; f32 into d_out)
    gemm_bf16_wmma<false, false, false><<<dim3(16, 32, 8), 256, 0, stream>>>(
        s1_b, TT, scT_b, TT, nullptr, scoreF, nullptr, TT, T, T, T, T, 1.0f);

    // score = a1*softmax(mixed) + (1-a1)*softmax(att); in-place on scoreF
    softmax_blend<<<dim3((unsigned)BT), dim3(256), 0, stream>>>(att_f, scoreF, sc_b, a1);

    // z = score @ v  (B = v^T [D][T1]; f32 out)  [8,2048,384]
    gemm_bf16_wmma<false, false, false><<<dim3(3, 32, 8), 256, 0, stream>>>(
        sc_b, TT, vT_b, TD, nullptr, z_f, nullptr, TD, T, D, T, D, 1.0f);

    // x = LN(tgt + z)
    add_ln<<<dim3((unsigned)BT), dim3(128), 0, stream>>>(tgt, z_f, g1, be1, x_f, x_b);

    // FFN: h = relu(x@W1+b1); ff = h@W2+b2
    gemm_bf16_wmma<true, true, false><<<dim3(12, 256, 1), 256, 0, stream>>>(
        x_b, 0, w1T, 0, b1, nullptr, h_b, 0, (int)BT, DF, D, DF, 1.0f);
    gemm_bf16_wmma<false, false, false><<<dim3(3, 256, 1), 256, 0, stream>>>(
        h_b, 0, w2T, 0, b2, ff_f, nullptr, 0, (int)BT, D, DF, D, 1.0f);

    // out = LN(x + ff)
    add_ln<<<dim3((unsigned)BT), dim3(128), 0, stream>>>(x_f, ff_f, g2, be2, outF, nullptr);
}
